// FeatureAlign_V2_4690104287796
// MI455X (gfx1250) — matrix operand order repaired and checked by reading, not compiled, not run on hardware
//
#include <hip/hip_runtime.h>
#include <cmath>

// ---------------------------------------------------------------------------
// FeatureAlign_V2 for MI455X (gfx1250, wave32, WMMA f32<-f16 16x16x32)
// ---------------------------------------------------------------------------
#define Bb 2
#define Cc 256
#define Hh 96
#define Ww 96
#define HW (Hh * Ww)        // 9216
#define Hs 48
#define Ws 48
#define Gg 8
#define Kk 9
#define OFFCH 144           // G*2*K
#define INNER_DCN (Cc * Kk) // 2304
#define PTILES (HW / 16)    // 576

typedef _Float16 v16h __attribute__((ext_vector_type(16)));
typedef float    v8f  __attribute__((ext_vector_type(8)));
typedef float    v4f  __attribute__((ext_vector_type(4)));

static __device__ __forceinline__ int imin(int a, int b) { return a < b ? a : b; }
static __device__ __forceinline__ int imax(int a, int b) { return a > b ? a : b; }

static __device__ __forceinline__ v8f wmma_f16(v16h a, v16h b, v8f c) {
  return __builtin_amdgcn_wmma_f32_16x16x32_f16(false, a, false, b, (short)0, c,
                                                false, false);
}

// A fragment: 16x32 f16 weights, row-major w[row*ldw + k].
// Lanes 0-15: M=lane, halves 0-7 -> K 0..7, halves 8-15 -> K 16..23.
// Lanes 16-31: M=lane-16, halves 0-7 -> K 8..15, halves 8-15 -> K 24..31.
static __device__ __forceinline__ v16h load_a(const _Float16* w, int row,
                                              int ldw, int kk, int lane) {
  const _Float16* p = w + row * ldw + kk + ((lane >> 4) << 3);
  v16h a;
  *(v4f*)&a       = *(const v4f*)p;        // halves 0..7
  *(((v4f*)&a)+1) = *(const v4f*)(p + 16); // halves 8..15
  return a;
}

// B fragment: 32x16 f16 activations staged in LDS as [pos][inner] (ld halves).
// Lanes 0-15: N=lane, halves 0-7 -> K 0..7, halves 8-15 -> K 16..23; lanes 16-31: K 8..15, K 24..31.
static __device__ __forceinline__ v16h load_b(const _Float16* smp, int ld,
                                              int kk, int lane) {
  const _Float16* p = smp + (lane & 15) * ld + kk + ((lane >> 4) << 3);
  v16h b;
  *(v4f*)&b       = *(const v4f*)p;       // halves 0..7
  *(((v4f*)&b)+1) = *(const v4f*)(p + 16); // halves 8..15
  return b;
}

// --------------------------- small helper kernels ---------------------------
__global__ void k_convert_f16(const float* __restrict__ x,
                              _Float16* __restrict__ y, int n) {
  int i = blockIdx.x * blockDim.x + threadIdx.x;
  if (i < n) y[i] = (_Float16)x[i];
}

__global__ void __launch_bounds__(256) k_mean(const float* __restrict__ feat_l,
                                              float* __restrict__ avg) {
  __shared__ float red[256];
  int bc = blockIdx.x;
  const float* p = feat_l + (size_t)bc * HW;
  float s = 0.f;
  for (int i = threadIdx.x; i < HW; i += 256) s += p[i];
  red[threadIdx.x] = s;
  __syncthreads();
  for (int off = 128; off > 0; off >>= 1) {
    if (threadIdx.x < off) red[threadIdx.x] += red[threadIdx.x + off];
    __syncthreads();
  }
  if (threadIdx.x == 0) avg[bc] = red[0] * (1.0f / HW);
}

__global__ void __launch_bounds__(256) k_atten(const float* __restrict__ atten_w,
                                               const float* __restrict__ avg,
                                               float* __restrict__ scale) {
  int b = blockIdx.x, o = threadIdx.x;
  const float* av = avg + b * Cc;
  float s = 0.f;
  for (int c = 0; c < Cc; ++c) s += atten_w[o * Cc + c] * av[c];
  scale[b * Cc + o] = 1.0f + 1.0f / (1.0f + __expf(-s));
}

// jax.image.resize bilinear 2x: half-pixel centers, edge clamp.
__global__ void __launch_bounds__(256) k_upsample(const float* __restrict__ feat_s,
                                                  float* __restrict__ feat_up) {
  int idx = blockIdx.x * 256 + threadIdx.x;
  if (idx >= Bb * Cc * HW) return;
  int p = idx % HW, bc = idx / HW;
  int oy = p / Ww, ox = p % Ww;
  float fy = oy * 0.5f - 0.25f, fx = ox * 0.5f - 0.25f;
  float y0f = floorf(fy), x0f = floorf(fx);
  float ty = fy - y0f, tx = fx - x0f;
  int y0 = (int)y0f, x0 = (int)x0f;
  int y0c = imin(imax(y0, 0), Hs - 1), y1c = imin(imax(y0 + 1, 0), Hs - 1);
  int x0c = imin(imax(x0, 0), Ws - 1), x1c = imin(imax(x0 + 1, 0), Ws - 1);
  const float* pl = feat_s + (size_t)bc * (Hs * Ws);
  float v = (1.f - ty) * ((1.f - tx) * pl[y0c * Ws + x0c] + tx * pl[y0c * Ws + x1c]) +
            ty         * ((1.f - tx) * pl[y1c * Ws + x0c] + tx * pl[y1c * Ws + x1c]);
  feat_up[idx] = v;
}

// ----------------------- feat_arm = fsm_w @ (x*scale) -----------------------
#define ARM_LDB 264
__global__ void __launch_bounds__(256) k_arm_gemm(const float* __restrict__ feat_l,
                                                  const float* __restrict__ scale,
                                                  const _Float16* __restrict__ w16,
                                                  float* __restrict__ feat_arm) {
  __shared__ _Float16 smp[16 * ARM_LDB];
  int b = blockIdx.y, pbase = blockIdx.x * 16, tid = threadIdx.x;
  int c = tid;  // 256 threads == 256 channels
  float sc = scale[b * Cc + c];
  const float* src = feat_l + ((size_t)(b * Cc + c)) * HW + pbase;
  for (int pos = 0; pos < 16; ++pos)
    smp[pos * ARM_LDB + c] = (_Float16)(src[pos] * sc);
  __syncthreads();

  int lane = tid & 31, wv = tid >> 5;
  int r0 = (wv * 2) * 16 + (lane & 15), r1 = r0 + 16;
  v8f acc0 = {}, acc1 = {};
  for (int kk = 0; kk < Cc; kk += 32) {
    v16h bm = load_b(smp, ARM_LDB, kk, lane);
    v16h a0 = load_a(w16, r0, Cc, kk, lane);
    v16h a1 = load_a(w16, r1, Cc, kk, lane);
    acc0 = wmma_f16(a0, bm, acc0);
    acc1 = wmma_f16(a1, bm, acc1);
  }
  int p = pbase + (lane & 15);
  int ob0 = (wv * 2) * 16 + ((lane >> 4) << 3), ob1 = ob0 + 16;
  for (int j = 0; j < 8; ++j) {
    feat_arm[((size_t)(b * Cc + ob0 + j)) * HW + p] = acc0[j];
    feat_arm[((size_t)(b * Cc + ob1 + j)) * HW + p] = acc1[j];
  }
}

// ------------------- offset = offset_w @ [feat_arm; 2*up] -------------------
#define OFF_LDB 520
__global__ void __launch_bounds__(288) k_off_gemm(const float* __restrict__ feat_arm,
                                                  const float* __restrict__ feat_up,
                                                  const _Float16* __restrict__ w16,
                                                  float* __restrict__ offs) {
  __shared__ _Float16 smp[16 * OFF_LDB];
  int b = blockIdx.y, pbase = blockIdx.x * 16, tid = threadIdx.x;
  for (int idx = tid; idx < 512 * 16; idx += 288) {
    int pos = idx & 15, inner = idx >> 4;
    int p = pbase + pos;
    float v = (inner < 256)
                ? feat_arm[((size_t)(b * Cc + inner)) * HW + p]
                : 2.0f * feat_up[((size_t)(b * Cc + inner - 256)) * HW + p];
    smp[pos * OFF_LDB + inner] = (_Float16)v;
  }
  __syncthreads();

  int lane = tid & 31, wv = tid >> 5;  // 9 waves == 9 M-tiles (144 rows)
  int r = wv * 16 + (lane & 15);
  v8f acc = {};
  for (int kk = 0; kk < 512; kk += 32) {
    v16h bm = load_b(smp, OFF_LDB, kk, lane);
    v16h a  = load_a(w16, r, 512, kk, lane);
    acc = wmma_f16(a, bm, acc);
  }
  int p = pbase + (lane & 15);
  int ob = wv * 16 + ((lane >> 4) << 3);
  for (int j = 0; j < 8; ++j)
    offs[((size_t)(b * OFFCH + ob + j)) * HW + p] = acc[j];
}

// ------------- deformable conv: sample -> LDS f16 -> WMMA GEMM --------------
struct Rec { float w00, w01, w10, w11; int i00, i01, i10, i11; };
#define DCN_LDB 2312
__global__ void __launch_bounds__(256) k_dcn(const float* __restrict__ feat_up,
                                             const float* __restrict__ offs,
                                             const _Float16* __restrict__ w16,
                                             const float* __restrict__ feat_arm,
                                             float* __restrict__ out) {
  __shared__ _Float16 smp[16 * DCN_LDB];  // 73,984 B
  __shared__ Rec rec[16 * Gg * Kk];       // 36,864 B
  int b = blockIdx.y, pbase = blockIdx.x * 16, tid = threadIdx.x;

  // Per-(pos, group, tap) bilinear weights (zero-pad masked) + clamped indices.
  for (int r = tid; r < 16 * 72; r += 256) {
    int pos = r / 72, gk = r % 72, k = gk % 9;
    int p = pbase + pos, ih = p / Ww, iw = p % Ww;
    float dy = offs[((size_t)(b * OFFCH + gk * 2)) * HW + p];
    float dx = offs[((size_t)(b * OFFCH + gk * 2 + 1)) * HW + p];
    float y = (float)(ih + k / 3 - 1) + dy;
    float x = (float)(iw + k % 3 - 1) + dx;
    float y0f = floorf(y), x0f = floorf(x);
    float ty = y - y0f, tx = x - x0f;
    int y0 = (int)y0f, x0 = (int)x0f, y1 = y0 + 1, x1 = x0 + 1;
    bool vy0 = (unsigned)y0 < (unsigned)Hh, vy1 = (unsigned)y1 < (unsigned)Hh;
    bool vx0 = (unsigned)x0 < (unsigned)Ww, vx1 = (unsigned)x1 < (unsigned)Ww;
    int cy0 = imin(imax(y0, 0), Hh - 1), cy1 = imin(imax(y1, 0), Hh - 1);
    int cx0 = imin(imax(x0, 0), Ww - 1), cx1 = imin(imax(x1, 0), Ww - 1);
    Rec rc;
    rc.w00 = (vy0 && vx0) ? (1.f - ty) * (1.f - tx) : 0.f;
    rc.w01 = (vy0 && vx1) ? (1.f - ty) * tx : 0.f;
    rc.w10 = (vy1 && vx0) ? ty * (1.f - tx) : 0.f;
    rc.w11 = (vy1 && vx1) ? ty * tx : 0.f;
    rc.i00 = cy0 * Ww + cx0; rc.i01 = cy0 * Ww + cx1;
    rc.i10 = cy1 * Ww + cx0; rc.i11 = cy1 * Ww + cx1;
    rec[r] = rc;
  }
  __syncthreads();

  // im2col fill (f16) for this block's 16 positions.
  for (int i = tid; i < INNER_DCN; i += 256) {
    int c = i / 9, k = i - c * 9, g = c >> 5;
    const float* pl = feat_up + ((size_t)(b * Cc + c)) * HW;
    int rb = g * 9 + k;
    for (int pos = 0; pos < 16; ++pos) {
      Rec rc = rec[pos * 72 + rb];
      float v = rc.w00 * pl[rc.i00] + rc.w01 * pl[rc.i01] +
                rc.w10 * pl[rc.i10] + rc.w11 * pl[rc.i11];
      smp[pos * DCN_LDB + i] = (_Float16)v;
    }
  }
  __syncthreads();

  int lane = tid & 31, wv = tid >> 5;
  int r0 = (wv * 2) * 16 + (lane & 15), r1 = r0 + 16;
  v8f acc0 = {}, acc1 = {};
  for (int kk = 0; kk < INNER_DCN; kk += 32) {
    v16h bm = load_b(smp, DCN_LDB, kk, lane);
    v16h a0 = load_a(w16, r0, INNER_DCN, kk, lane);
    v16h a1 = load_a(w16, r1, INNER_DCN, kk, lane);
    acc0 = wmma_f16(a0, bm, acc0);
    acc1 = wmma_f16(a1, bm, acc1);
  }
  int p = pbase + (lane & 15);
  int ob0 = (wv * 2) * 16 + ((lane >> 4) << 3), ob1 = ob0 + 16;
  for (int j = 0; j < 8; ++j) {
    size_t i0 = ((size_t)(b * Cc + ob0 + j)) * HW + p;
    size_t i1 = ((size_t)(b * Cc + ob1 + j)) * HW + p;
    float v0 = acc0[j]; v0 = v0 > 0.f ? v0 : 0.f;
    float v1 = acc1[j]; v1 = v1 > 0.f ? v1 : 0.f;
    out[i0] = v0 + feat_arm[i0];
    out[i1] = v1 + feat_arm[i1];
  }
}

// ------------------------------- launcher -----------------------------------
extern "C" void kernel_launch(void* const* d_in, const int* in_sizes, int n_in,
                              void* d_out, int out_size, void* d_ws, size_t ws_size,
                              hipStream_t stream) {
  const float* feat_l   = (const float*)d_in[0];
  const float* feat_s   = (const float*)d_in[1];
  const float* atten_w  = (const float*)d_in[2];
  const float* fsm_w    = (const float*)d_in[3];
  const float* offset_w = (const float*)d_in[4];
  const float* dcn_w    = (const float*)d_in[5];
  float* out = (float*)d_out;

  char* w = (char*)d_ws;
  float* avg      = (float*)w;  w += sizeof(float) * Bb * Cc;
  float* scalep   = (float*)w;  w += sizeof(float) * Bb * Cc;
  float* feat_up  = (float*)w;  w += sizeof(float) * (size_t)Bb * Cc * HW;
  float* feat_arm = (float*)w;  w += sizeof(float) * (size_t)Bb * Cc * HW;
  float* offs     = (float*)w;  w += sizeof(float) * (size_t)Bb * OFFCH * HW;
  _Float16* w16_fsm = (_Float16*)w;  w += sizeof(_Float16) * Cc * Cc;
  _Float16* w16_off = (_Float16*)w;  w += sizeof(_Float16) * OFFCH * 2 * Cc;
  _Float16* w16_dcn = (_Float16*)w;  w += sizeof(_Float16) * Cc * INNER_DCN;

  k_convert_f16<<<(Cc * Cc + 255) / 256, 256, 0, stream>>>(fsm_w, w16_fsm, Cc * Cc);
  k_convert_f16<<<(OFFCH * 2 * Cc + 255) / 256, 256, 0, stream>>>(offset_w, w16_off, OFFCH * 2 * Cc);
  k_convert_f16<<<(Cc * INNER_DCN + 255) / 256, 256, 0, stream>>>(dcn_w, w16_dcn, Cc * INNER_DCN);

  k_mean<<<Bb * Cc, 256, 0, stream>>>(feat_l, avg);
  k_atten<<<Bb, 256, 0, stream>>>(atten_w, avg, scalep);
  k_upsample<<<(Bb * Cc * HW + 255) / 256, 256, 0, stream>>>(feat_s, feat_up);

  k_arm_gemm<<<dim3(PTILES, Bb), 256, 0, stream>>>(feat_l, scalep, w16_fsm, feat_arm);
  k_off_gemm<<<dim3(PTILES, Bb), 288, 0, stream>>>(feat_arm, feat_up, w16_off, offs);
  k_dcn<<<dim3(PTILES, Bb), 256, 0, stream>>>(feat_up, offs, w16_dcn, feat_arm, out);
}
